// ExampleGuidedAttention_41918880809207
// MI455X (gfx1250) — compile-verified, matrix operand order checked by reading, not run on hardware
//
#include <hip/hip_runtime.h>

// ---------------------------------------------------------------------------
// Guided attention, fused flash-attention style for gfx1250 (MI455X).
//   N=8, C=256, Cq=64, H=W=64, HW=4096.
//   Q = conv(src)            -> bf16 [n][i][64]      (ws)
//   V = concat(src, ref)     -> bf16 [n][col512][j]  (ws)
//   O[n, i, col] = softmax_j(Q_i . Q_j) @ V  (flash, never materialized)
//   epilogue: mask blend + channel concat to fp32 output.
// Round 2: async K-tile double buffering (GLOBAL_LOAD_ASYNC_TO_LDS_B128 +
// s_wait_asynccnt) and fully parallel shuffle-based online softmax.
// ---------------------------------------------------------------------------

typedef __attribute__((ext_vector_type(16))) __bf16         v16bf;
typedef __attribute__((ext_vector_type(8)))  float          v8f;
typedef __attribute__((ext_vector_type(16))) unsigned short v16u;
typedef __attribute__((ext_vector_type(8)))  unsigned short v8u;

#define GA_N   8
#define GA_C   256
#define GA_CQ  64
#define GA_HW  4096
#define GA_BM  64          // query rows per block
#define GA_BN  64          // kv columns per tile
#define GA_V   512         // concatenated V channels (src 256 + ref 256)
#define GA_NT  (GA_HW / GA_BN)
#define WAVES  16
#define TPB    (WAVES * 32)

// float -> bf16 round-to-nearest-even (bit math; no __bf16 arithmetic needed)
__device__ __forceinline__ unsigned short f2bf(float f) {
  union { float f; unsigned u; } v; v.f = f;
  unsigned u = v.u;
  return (unsigned short)((u + 0x7FFFu + ((u >> 16) & 1u)) >> 16);
}

// Async DMA: 16 bytes per lane, global (SADDR base + 32-bit voffset) -> LDS.
// Tracked by ASYNCcnt; LDS address = low 32 bits of generic pointer (ISA 10.2).
__device__ __forceinline__ void async_load_b128(unsigned lds_addr,
                                                const unsigned short* gbase,
                                                unsigned voff_bytes) {
  asm volatile("global_load_async_to_lds_b128 %0, %1, %2"
               :
               : "v"(lds_addr), "v"(voff_bytes), "s"(gbase)
               : "memory");
}
__device__ __forceinline__ void wait_async0() {
  asm volatile("s_wait_asynccnt 0" ::: "memory");
}

// Load one 16x32 bf16 WMMA A/B fragment from a row-major [row][k] tile where
// the k (reduction) dimension is contiguous. Per CDNA5 ISA 7.12.2 layout:
// lane = 16*g + r holds k = {k0+8g .. k0+8g+7} and {k0+16+8g .. k0+23+8g}
// of row r  -> two 16-byte contiguous chunks per lane.
__device__ __forceinline__ v16bf load_frag(const unsigned short* tile,
                                           int pitch, int k0, int lane) {
  const int r = lane & 15, g = lane >> 4;
  const unsigned short* p = tile + (size_t)r * pitch + k0 + g * 8;
  v8u c0 = *(const v8u*)(p);
  v8u c1 = *(const v8u*)(p + 16);
  v16u o;
#pragma unroll
  for (int e = 0; e < 8; ++e) { o[e] = c0[e]; o[e + 8] = c1[e]; }
  return __builtin_bit_cast(v16bf, o);
}

// ---------------- prep 1: query GEMM (tiny: 1.07 GFLOP) --------------------
__global__ __launch_bounds__(256)
void ga_make_query(const float* __restrict__ src_feature,
                   const float* __restrict__ conv_w,
                   unsigned short* __restrict__ Qb) {
  const int oc = threadIdx.x;                       // 0..63
  const int ig = blockIdx.x;                        // n*1024 + pixel-group
  const int n  = ig >> 10;
  const int i  = ((ig & 1023) << 2) + threadIdx.y;  // pixel
  const float* src = src_feature + (size_t)n * GA_C * GA_HW + i;
  const float* w   = conv_w + oc * GA_C;
  float s = 0.0f;
#pragma unroll 4
  for (int c = 0; c < GA_C; ++c) s = fmaf(w[c], src[(size_t)c * GA_HW], s);
  Qb[((size_t)n * GA_HW + i) * GA_CQ + oc] = f2bf(s);
}

// ---------------- prep 2: bf16 convert of concatenated V -------------------
__global__ __launch_bounds__(256)
void ga_make_v(const float* __restrict__ src_feature,
               const float* __restrict__ ref_feature,
               unsigned short* __restrict__ Vb) {
  size_t idx = (size_t)blockIdx.x * 256 + threadIdx.x;   // N*512*HW
  int j   = (int)(idx & (GA_HW - 1));
  int col = (int)((idx >> 12) & (GA_V - 1));
  int n   = (int)(idx >> 21);
  float v = (col < GA_C)
      ? src_feature[((size_t)n * GA_C + col) * GA_HW + j]
      : ref_feature[((size_t)n * GA_C + (col - GA_C)) * GA_HW + j];
  Vb[idx] = f2bf(v);
}

// ---------------- main fused flash-attention kernel ------------------------
__global__ __launch_bounds__(TPB)
void ga_flash_attn(const unsigned short* __restrict__ Qb,
                   const unsigned short* __restrict__ Vb,
                   const float* __restrict__ src_mask,
                   const float* __restrict__ ref_feature,
                   float* __restrict__ out) {
  __shared__ unsigned short Qs[GA_BM][GA_CQ];          // 8 KB
  __shared__ unsigned short Ks[2][GA_BN][GA_CQ];       // 16 KB (double buffer)
  __shared__ unsigned short Ps[GA_BM][GA_BN];          // 8 KB
  __shared__ float          Ss[GA_BM][GA_BN + 4];      // 17 KB (16B-aligned pad)
  __shared__ float mrow[GA_BM], lrow[GA_BM], arow[GA_BM];

  const int tid  = threadIdx.x;
  const int wave = tid >> 5;
  const int lane = tid & 31;
  const int n    = blockIdx.x >> 6;                    // HW/BM = 64 tiles/batch
  const int i0   = (blockIdx.x & 63) * GA_BM;

  const unsigned short* Qn = Qb + (size_t)n * GA_HW * GA_CQ;
  const unsigned short* Vn = Vb + (size_t)n * GA_V * GA_HW;

  // per-thread 16-byte chunk of a 64x64 bf16 tile
  const int krow = tid >> 3;
  const int kch  = (tid & 7) * 8;

  // Q tile -> LDS (one-time, synchronous)
  *(v8u*)&Qs[krow][kch] = *(const v8u*)(Qn + (size_t)(i0 + krow) * GA_CQ + kch);
  // kick off async DMA of K tile 0 into buffer 0
  async_load_b128((unsigned)(uintptr_t)&Ks[0][krow][kch], Qn,
                  (unsigned)(((unsigned)krow * GA_CQ + kch) * 2));
  if (tid < GA_BM) { mrow[tid] = -3.0e38f; lrow[tid] = 0.0f; }

  v8f acc[4][2];
  const v8f vzero = {};
#pragma unroll
  for (int mt = 0; mt < 4; ++mt)
#pragma unroll
    for (int nt = 0; nt < 2; ++nt) acc[mt][nt] = vzero;

  const int smt = wave >> 2;        // this wave's S tile (16 waves = 4x4 tiles)
  const int snt = wave & 3;
  const int colbase = wave * 32;    // this wave's 32-channel slice of V

  for (int t = 0; t < GA_NT; ++t) {
    const int cur = t & 1;
    const int j0  = t * GA_BN;

    // our async chunk of Ks[cur] has landed; publish to the workgroup
    wait_async0();
    __syncthreads();

    // overlap: DMA the next K tile into the other buffer during compute
    if (t + 1 < GA_NT)
      async_load_b128((unsigned)(uintptr_t)&Ks[cur ^ 1][krow][kch], Qn,
                      (unsigned)((((unsigned)(j0 + GA_BN) + krow) * GA_CQ + kch) * 2));

    // S = Q K^T : one 16x16 tile per wave, K-dim 64 -> 2 WMMAs
    {
      v8f s = vzero;
#pragma unroll
      for (int kk = 0; kk < GA_CQ; kk += 32) {
        v16bf a = load_frag(&Qs[smt * 16][0], GA_CQ, kk, lane);
        v16bf b = load_frag(&Ks[cur][snt * 16][0], GA_CQ, kk, lane);
        s = __builtin_amdgcn_wmma_f32_16x16x32_bf16(false, a, false, b,
                                                    (short)0, s, false, false);
      }
#pragma unroll
      for (int r = 0; r < 8; ++r)
        Ss[smt * 16 + r + ((lane >> 4) << 3)][snt * 16 + (lane & 15)] = s[r];
    }
    __syncthreads();

    // parallel online softmax: 8 lanes per row (wave-aligned groups),
    // shuffle reductions, no extra barriers
    {
      const int row = tid >> 3;
      const int seg = tid & 7;
      const float4* sp = (const float4*)&Ss[row][seg * 8];
      float4 s0 = sp[0], s1 = sp[1];
      float e[8] = {s0.x, s0.y, s0.z, s0.w, s1.x, s1.y, s1.z, s1.w};
      float mx = e[0];
#pragma unroll
      for (int k = 1; k < 8; ++k) mx = fmaxf(mx, e[k]);
      mx = fmaxf(mx, __shfl_xor(mx, 1));
      mx = fmaxf(mx, __shfl_xor(mx, 2));
      mx = fmaxf(mx, __shfl_xor(mx, 4));
      const float mold = mrow[row];
      const float mnew = fmaxf(mold, mx);
      float sum = 0.0f;
      v8u pv;
#pragma unroll
      for (int k = 0; k < 8; ++k) {
        float p = __expf(e[k] - mnew);
        sum += p;
        pv[k] = f2bf(p);
      }
      *(v8u*)&Ps[row][seg * 8] = pv;
      sum += __shfl_xor(sum, 1);
      sum += __shfl_xor(sum, 2);
      sum += __shfl_xor(sum, 4);
      if (seg == 0) {
        float alpha = __expf(mold - mnew);
        mrow[row] = mnew;
        lrow[row] = lrow[row] * alpha + sum;
        arow[row] = alpha;
      }
    }
    __syncthreads();

    // rescale accumulators by alpha, then O += P @ V (V streamed from L2)
#pragma unroll
    for (int mt = 0; mt < 4; ++mt)
#pragma unroll
      for (int nt = 0; nt < 2; ++nt)
#pragma unroll
        for (int r = 0; r < 8; ++r)
          acc[mt][nt][r] *= arow[mt * 16 + r + ((lane >> 4) << 3)];

#pragma unroll
    for (int mt = 0; mt < 4; ++mt) {
      v16bf a0 = load_frag(&Ps[mt * 16][0], GA_BN, 0, lane);
      v16bf a1 = load_frag(&Ps[mt * 16][0], GA_BN, 32, lane);
#pragma unroll
      for (int nt = 0; nt < 2; ++nt) {
        const unsigned short* vt = Vn + (size_t)(colbase + nt * 16) * GA_HW + j0;
        if (t + 1 < GA_NT)
          __builtin_prefetch(vt + (size_t)(lane & 15) * GA_HW + GA_BN, 0, 1);
        v16bf b0 = load_frag(vt, GA_HW, 0, lane);
        v16bf b1 = load_frag(vt, GA_HW, 32, lane);
        acc[mt][nt] = __builtin_amdgcn_wmma_f32_16x16x32_bf16(
            false, a0, false, b0, (short)0, acc[mt][nt], false, false);
        acc[mt][nt] = __builtin_amdgcn_wmma_f32_16x16x32_bf16(
            false, a1, false, b1, (short)0, acc[mt][nt], false, false);
      }
    }
  }

  __syncthreads();

  // epilogue: normalize, mask-blend, channel concat
#pragma unroll
  for (int mt = 0; mt < 4; ++mt) {
#pragma unroll
    for (int nt = 0; nt < 2; ++nt) {
      int col = colbase + nt * 16 + (lane & 15);
#pragma unroll
      for (int r = 0; r < 8; ++r) {
        int row = mt * 16 + r + ((lane >> 4) << 3);
        int i   = i0 + row;
        float val = acc[mt][nt][r] / lrow[row];
        if (col < GA_C) {
          // src attention -> second half of output channels
          out[((size_t)n * 2 * GA_C + GA_C + col) * GA_HW + i] = val;
        } else {
          int c = col - GA_C;
          float mval = src_mask[(size_t)n * GA_HW + i];
          float rf   = ref_feature[((size_t)n * GA_C + c) * GA_HW + i];
          out[((size_t)n * 2 * GA_C + c) * GA_HW + i] =
              (1.0f - mval) * val + mval * rf;
        }
      }
    }
  }
}

// ---------------------------------------------------------------------------
extern "C" void kernel_launch(void* const* d_in, const int* in_sizes, int n_in,
                              void* d_out, int out_size, void* d_ws, size_t ws_size,
                              hipStream_t stream) {
  const float* src_mask    = (const float*)d_in[0];
  const float* src_feature = (const float*)d_in[1];
  const float* ref_feature = (const float*)d_in[2];
  const float* conv_w      = (const float*)d_in[3];
  float* out = (float*)d_out;

  // workspace: Qb bf16 [8][4096][64] = 4 MB ; Vb bf16 [8][512][4096] = 32 MB
  unsigned short* Qb = (unsigned short*)d_ws;
  unsigned short* Vb = Qb + (size_t)GA_N * GA_HW * GA_CQ;

  ga_make_query<<<dim3(GA_N * (GA_HW / 4)), dim3(64, 4), 0, stream>>>(
      src_feature, conv_w, Qb);
  ga_make_v<<<dim3((GA_N * GA_V * GA_HW) / 256), dim3(256), 0, stream>>>(
      src_feature, ref_feature, Vb);
  ga_flash_attn<<<dim3(GA_N * (GA_HW / GA_BM)), dim3(TPB), 0, stream>>>(
      Qb, Vb, src_mask, ref_feature, out);
}
